// BinOverlapPredictionFromProjModule_85023172592446
// MI455X (gfx1250) — matrix operand order repaired and checked by reading, not compiled, not run on hardware
//
#include <hip/hip_runtime.h>
#include <hip/hip_bf16.h>
#include <math.h>

typedef __attribute__((ext_vector_type(16))) __bf16 v16bf;
typedef __attribute__((ext_vector_type(8)))  __bf16 v8bf;
typedef __attribute__((ext_vector_type(8)))  float  v8f;

#define THREADS 256

// ---------------- LDS layout (bytes) ----------------
// Activation buffers are padded to a multiple of 32 pixels so WMMA result
// stores need no bounds checks (slack space is never read back).
static constexpr int OFF_W   = 0;        // weights, max 576*64*2       = 73728
static constexpr int OFF_A1  = 73728;    // conv1 out 928px*64 bf16    = 118784
static constexpr int OFF_IN  = 192512;   // input 32*32*16 bf16 (32768), reused as pool 15*15*64 (28800)
static constexpr int OFF_A2  = 225280;   // conv2 out 192px*64 bf16    = 24576
static constexpr int OFF_A3  = 249856;   // conv3 out 128px*64 bf16    = 16384
static constexpr int OFF_RED = 266240;   // 256 f32 (reductions, then per-layer scale[64]+bias[64])
static constexpr int OFF_ARR = 267264;   // 376 f32 (ptm_max/weights/scalars/partials)
static constexpr int OFF_S4  = 268768;   // 96 f32 conv4 output (81 valid)
static constexpr int SMEM_BYTES = 269184;

// ---------------- workspace layout (bytes) ----------------
//  W1 [64][160] bf16 @ 0      (20480)
//  W2 [64][576] bf16 @ 20480  (73728)
//  W3 [64][576] bf16 @ 94208  (73728)
//  W4 [16][576] bf16 @ 167936 (18432)  (row 0 real, rows 1..15 zero)
//  SB f32 @ 186368: {s1[64],b1[64],s2[64],b2[64],s3[64],b3[64]}

// ======================================================================
// Weight transform + BN folding
// ======================================================================
__global__ __launch_bounds__(THREADS) void prep_kernel(
    const float* __restrict__ c1w, const float* __restrict__ c2w,
    const float* __restrict__ c3w, const float* __restrict__ c4w,
    const float* __restrict__ c1b, const float* __restrict__ c2b, const float* __restrict__ c3b,
    const float* __restrict__ g1, const float* __restrict__ b1, const float* __restrict__ m1, const float* __restrict__ v1,
    const float* __restrict__ g2, const float* __restrict__ b2, const float* __restrict__ m2, const float* __restrict__ v2,
    const float* __restrict__ g3, const float* __restrict__ b3, const float* __restrict__ m3, const float* __restrict__ v3,
    unsigned short* __restrict__ w1u, unsigned short* __restrict__ w2u,
    unsigned short* __restrict__ w3u, unsigned short* __restrict__ w4u,
    float* __restrict__ sb)
{
  __bf16* w1 = (__bf16*)w1u;
  __bf16* w2 = (__bf16*)w2u;
  __bf16* w3 = (__bf16*)w3u;
  __bf16* w4 = (__bf16*)w4u;
  const int id = blockIdx.x * THREADS + threadIdx.x;
  if (id < 10240) {                       // W1: [o][k], k = tap*16 + ci, K padded 144->160
    const int o = id / 160, k = id % 160;
    const int tap = k >> 4, ci = k & 15;
    float v = 0.f;
    if (tap < 9 && ci < 14) v = c1w[o * 126 + ci * 9 + tap];
    w1[id] = (__bf16)v;
  } else if (id < 47104) {                // W2: [o][k], k = tap*64 + ci
    const int i = id - 10240;
    const int o = i / 576, k = i % 576;
    const int tap = k >> 6, ci = k & 63;
    w2[i] = (__bf16)c2w[o * 576 + ci * 9 + tap];
  } else if (id < 83968) {                // W3
    const int i = id - 47104;
    const int o = i / 576, k = i % 576;
    const int tap = k >> 6, ci = k & 63;
    w3[i] = (__bf16)c3w[o * 576 + ci * 9 + tap];
  } else if (id < 93184) {                // W4: [16][576], only o==0 real
    const int i = id - 83968;
    const int o = i / 576, k = i % 576;
    const int tap = k >> 6, ci = k & 63;
    const float v = (o == 0) ? c4w[ci * 9 + tap] : 0.f;
    w4[i] = (__bf16)v;
  } else if (id < 93376) {                // fold BN: y = s*conv + (s*(cb-m)+bb)
    const int i = id - 93184;
    const int l = i >> 6, c = i & 63;
    const float *g, *bb, *mm, *vv, *cb;
    if (l == 0)      { g = g1; bb = b1; mm = m1; vv = v1; cb = c1b; }
    else if (l == 1) { g = g2; bb = b2; mm = m2; vv = v2; cb = c2b; }
    else             { g = g3; bb = b3; mm = m3; vv = v3; cb = c3b; }
    const float s = g[c] * rsqrtf(vv[c] + 1e-5f);
    sb[l * 128 + c]      = s;
    sb[l * 128 + 64 + c] = bb[c] + s * (cb[c] - mm[c]);
  }
}

// ======================================================================
// LDS helpers
// ======================================================================
__device__ __forceinline__ void copy_lds(void* dst, const void* src, int bytes, int t) {
  const uint4* s = (const uint4*)src;
  uint4* d = (uint4*)dst;
  const int n = bytes >> 4;
  for (int i = t; i < n; i += THREADS) d[i] = s[i];
}

// ======================================================================
// A-fragment gather for one 16x32 bf16 tile row set.
// lane = {row, hi}; elements 0..7 = K base+hi*8+(0..7),
// elements 8..15 = K base+16+hi*8+(0..7)  -> two 16B ds loads.
// ======================================================================
template<int Ws, int Cin>
__device__ __forceinline__ v16bf gather_a(const __bf16* __restrict__ src,
                                          int py, int px, int hi, int kc)
{
  union { v16bf v; v8bf h[2]; } ua;
  if constexpr (Cin == 64) {
    const int tap = kc >> 1;
    const int ky = tap / 3, kx = tap % 3;
    const int ci0 = ((kc & 1) << 5) + hi * 8;
    const __bf16* p = src + ((py + ky) * Ws + (px + kx)) * 64 + ci0;
    ua.h[0] = *(const v8bf*)p;
    ua.h[1] = *(const v8bf*)(p + 16);
  } else {                                // Cin == 16 (conv1): 2 taps per K-chunk
    const int tap0 = kc * 2, tap1 = kc * 2 + 1;
    {
      const int ky = tap0 / 3, kx = tap0 % 3;
      ua.h[0] = *(const v8bf*)(src + ((py + ky) * Ws + (px + kx)) * 16 + hi * 8);
    }
    if (tap1 < 9) {
      const int ky = tap1 / 3, kx = tap1 % 3;
      ua.h[1] = *(const v8bf*)(src + ((py + ky) * Ws + (px + kx)) * 16 + hi * 8);
    } else {
      v8bf z = {};
      ua.h[1] = z;                        // K padding (weights are zero there too)
    }
  }
  return ua.v;
}

// ======================================================================
// Implicit-GEMM conv layer on v_wmma_f32_16x16x32_bf16, dual-M blocked:
// each wave handles two adjacent 16-pixel M tiles per N tile, so every
// B fragment feeds two independent WMMA chains.
//   src: LDS activations [H][W][Cin] bf16 (channel-contig)
//   sw : LDS weights [Cout][KP] bf16
//   dst: LDS [>= ceil(NPIX/32)*32 pixels][64] bf16 (padded, unguarded stores)
// ======================================================================
template<int Ws, int Wd, int NPIX, int Cin, int KP, int NT, bool RELU, bool FINAL>
__device__ __forceinline__ void conv_wmma(const __bf16* __restrict__ src,
                                          __bf16* __restrict__ dst,
                                          const __bf16* __restrict__ sw,
                                          const float* __restrict__ ssb,
                                          float* __restrict__ s4out,
                                          int lane, int wave)
{
  constexpr int MT  = (NPIX + 15) / 16;
  constexpr int MT2 = (MT + 1) / 2;
  constexpr int KC  = KP / 32;
  const int row = lane & 15;
  const int hi  = lane >> 4;
  for (int tc = wave; tc < MT2 * NT; tc += 8) {
    const int mt = (tc / NT) * 2, nt = tc % NT;
    const int mp0 = mt * 16;
    int pix0 = mp0 + row;      if (pix0 > NPIX - 1) pix0 = NPIX - 1;
    int pix1 = mp0 + 16 + row; if (pix1 > NPIX - 1) pix1 = NPIX - 1;
    const int py0 = pix0 / Wd, px0 = pix0 % Wd;
    const int py1 = pix1 / Wd, px1 = pix1 % Wd;
    const __bf16* bcol = sw + (nt * 16 + row) * KP;
    v8f acc0 = {}, acc1 = {};
#pragma unroll
    for (int kc = 0; kc < KC; ++kc) {
      const v16bf a0 = gather_a<Ws, Cin>(src, py0, px0, hi, kc);
      const v16bf a1 = gather_a<Ws, Cin>(src, py1, px1, hi, kc);
      const v16bf bw = __builtin_shufflevector(*(const v8bf*)(bcol + kc * 32 + hi * 8),
                                               *(const v8bf*)(bcol + kc * 32 + 16 + hi * 8),
                                               0, 1, 2, 3, 4, 5, 6, 7, 8, 9, 10, 11, 12, 13, 14, 15);
      acc0 = __builtin_amdgcn_wmma_f32_16x16x32_bf16(false, a0, false, bw,
                                                     (short)0, acc0, false, false);
      acc1 = __builtin_amdgcn_wmma_f32_16x16x32_bf16(false, a1, false, bw,
                                                     (short)0, acc1, false, false);
    }
    if constexpr (!FINAL) {
      const int co = nt * 16 + row;
      const float sc = ssb[co], bi = ssb[64 + co];
      __bf16* d0 = dst + (mp0 + hi * 8) * 64 + co;
#pragma unroll
      for (int r = 0; r < 8; ++r) {       // element r -> pixel mp0 + hi*8 + r (tile0)
        float v = acc0[r] * sc + bi;
        if (RELU) v = fmaxf(v, 0.0f);
        d0[r * 64] = (__bf16)v;
      }
      __bf16* d1 = d0 + 16 * 64;
#pragma unroll
      for (int r = 0; r < 8; ++r) {       // tile1
        float v = acc1[r] * sc + bi;
        if (RELU) v = fmaxf(v, 0.0f);
        d1[r * 64] = (__bf16)v;
      }
    } else {
      if (row == 0) {                     // only column 0 is the real output channel
        float* q0 = s4out + mp0 + hi * 8;
#pragma unroll
        for (int r = 0; r < 8; ++r) q0[r] = acc0[r];
        float* q1 = q0 + 16;
#pragma unroll
        for (int r = 0; r < 8; ++r) q1[r] = acc1[r];
      }
    }
  }
}

// ======================================================================
// Fully fused per-sample kernel: stats -> input build -> conv1 -> pool
// -> conv2 -> conv3 -> conv4 -> global max. One workgroup per sample.
// ======================================================================
__global__ __launch_bounds__(THREADS, 1) void fused_kernel(
    const float* __restrict__ ts, const float* __restrict__ ptm,
    const float* __restrict__ pmt, const float* __restrict__ piou,
    const int* __restrict__ mmask,
    const unsigned short* __restrict__ gW1u, const unsigned short* __restrict__ gW2u,
    const unsigned short* __restrict__ gW3u, const unsigned short* __restrict__ gW4u,
    const float* __restrict__ gSB, const float* __restrict__ c4b,
    float* __restrict__ out)
{
  extern __shared__ char smem[];
  __bf16* sW    = (__bf16*)(smem + OFF_W);
  __bf16* sA1   = (__bf16*)(smem + OFF_A1);
  __bf16* sIN   = (__bf16*)(smem + OFF_IN);
  __bf16* sPool = (__bf16*)(smem + OFF_IN);     // reuse: input slab dead after conv1
  __bf16* sA2   = (__bf16*)(smem + OFF_A2);
  __bf16* sA3   = (__bf16*)(smem + OFF_A3);
  float*  sRed  = (float*)(smem + OFF_RED);     // 256 f32; later per-layer scale/bias
  float*  sArr  = (float*)(smem + OFF_ARR);
  float*  sPtmMax = sArr;                       // 30
  float*  sWm   = sArr + 30;                    // 30
  float*  sWg   = sArr + 60;                    // 30
  float*  sWo   = sArr + 90;                    // 30
  float*  sScal = sArr + 120;                   // 16
  float*  sPart = sArr + 136;                   // 240
  float*  sS4   = (float*)(smem + OFF_S4);

  const __bf16* gW1 = (const __bf16*)gW1u;
  const __bf16* gW2 = (const __bf16*)gW2u;
  const __bf16* gW3 = (const __bf16*)gW3u;
  const __bf16* gW4 = (const __bf16*)gW4u;

  const int b = blockIdx.x;
  const int t = threadIdx.x;
  const int lane = t & 31, wave = t >> 5;
  const size_t pbase = (size_t)b * 30 * 1024;

  __builtin_prefetch(pmt + pbase, 0, 1);        // global_prefetch_b8
  __builtin_prefetch(ptm + pbase, 0, 1);

  // ---- phase 0a: mask weights (mf, gth, oth) and counts ----
  if (t == 0) {
    float cum = 0.f, cnt = 0.f, cntg = 0.f, cnto = 0.f;
    for (int m = 0; m < 30; ++m) {
      const float mf = mmask[b * 30 + m] ? 1.0f : 0.0f;
      cum += mf;
      const float rank = cum - 1.0f;
      const float g = (mf > 0.f && rank < 15.f) ? 1.0f : 0.0f;
      const float o = (mf > 0.f && rank >= 15.f) ? 1.0f : 0.0f;
      sWm[m] = mf; sWg[m] = g; sWo[m] = o;
      cnt += mf; cntg += g; cnto += o;
    }
    sScal[7] = cnt;
    sScal[8] = (cnto > 0.f) ? cnto : 1.0f;
    sScal[9] = (cntg > 0.f) ? cntg : 1.0f;
    sScal[10] = piou[b];
  }
  // ---- phase 0b: max over target_scores + per-mem max over ptm ----
  {
    const float* tb = ts + (size_t)b * 1024;
    float mx = -INFINITY;
    for (int j = 0; j < 4; ++j) mx = fmaxf(mx, tb[t + j * 256]);
    sRed[t] = mx;
  }
  if (t < 240) {                                 // 30 mems x 8 chunks of 128 pixels
    const int m = t >> 3, ch = t & 7;
    const float* p = ptm + pbase + m * 1024 + ch * 128;
    float mx = -INFINITY;
    for (int i = 0; i < 128; ++i) mx = fmaxf(mx, p[i]);
    sPart[t] = mx;
  }
  __syncthreads();
  for (int s = 128; s > 0; s >>= 1) {
    if (t < s) sRed[t] = fmaxf(sRed[t], sRed[t + s]);
    __syncthreads();
  }
  if (t < 30) {
    float mx = -INFINITY;
    for (int c = 0; c < 8; ++c) mx = fmaxf(mx, sPart[t * 8 + c]);
    sPtmMax[t] = mx;
  }
  __syncthreads();
  if (t == 0) {
    sScal[0] = sRed[0];                          // max_ts
    const float cnt = sScal[7], cnto = sScal[8], cntg = sScal[9];
    float sm = 0, sm2 = 0, sg = 0, sg2 = 0, so = 0, so2 = 0;
    for (int m = 0; m < 30; ++m) {
      const float v = sPtmMax[m];
      sm += sWm[m] * v;  sm2 += sWm[m] * v * v;
      sg += sWg[m] * v;  sg2 += sWg[m] * v * v;
      so += sWo[m] * v;  so2 += sWo[m] * v * v;
    }
    const float mm = sm / cnt;   sScal[1] = mm; sScal[2] = sqrtf(fmaxf(sm2 / cnt  - mm * mm, 0.f));
    const float mg = sg / cntg;  sScal[3] = mg; sScal[4] = sqrtf(fmaxf(sg2 / cntg - mg * mg, 0.f));
    const float mo = so / cnto;  sScal[5] = mo; sScal[6] = sqrtf(fmaxf(so2 / cnto - mo * mo, 0.f));
  }
  __syncthreads();

  // ---- phase 0c: per-pixel masked pmt stats -> build (32,32,16) bf16 input slab ----
  {
    float aS[4] = {0, 0, 0, 0}, aSS[4] = {0, 0, 0, 0};
    float gS[4] = {0, 0, 0, 0}, gSS[4] = {0, 0, 0, 0};
    float oS[4] = {0, 0, 0, 0}, oSS[4] = {0, 0, 0, 0};
    const float* pm = pmt + pbase;
    for (int m = 0; m < 30; ++m) {
      const float wm = sWm[m], wo = sWo[m];
      const float* pp = pm + m * 1024 + t;
#pragma unroll
      for (int j = 0; j < 4; ++j) {
        const float x = pp[j * 256];
        aS[j] += wm * x; aSS[j] += wm * x * x;
        oS[j] += wo * x; oSS[j] += wo * x * x;
        if (m < 15) { gS[j] += x; gSS[j] += x * x; }
      }
    }
    const float cnt = sScal[7], cnto = sScal[8];
    const float c0 = sScal[0], c1 = sScal[10];
    const float c8 = sScal[1], c9 = sScal[2], c10 = sScal[3];
    const float c11 = sScal[4], c12 = sScal[5], c13 = sScal[6];
#pragma unroll
    for (int j = 0; j < 4; ++j) {
      const int p = t + j * 256;
      __bf16* o = sIN + p * 16;
      const float mean = aS[j] / cnt;
      const float stdv = sqrtf(fmaxf(aSS[j] / cnt - mean * mean, 0.f));
      const float gme = gS[j] * (1.0f / 15.0f);
      const float gst = sqrtf(fmaxf(gSS[j] * (1.0f / 15.0f) - gme * gme, 0.f));
      const float ome = oS[j] / cnto;
      const float ost = sqrtf(fmaxf(oSS[j] / cnto - ome * ome, 0.f));
      o[0] = (__bf16)c0;   o[1] = (__bf16)c1;   o[2] = (__bf16)mean; o[3] = (__bf16)stdv;
      o[4] = (__bf16)gme;  o[5] = (__bf16)gst;  o[6] = (__bf16)ome;  o[7] = (__bf16)ost;
      o[8] = (__bf16)c8;   o[9] = (__bf16)c9;   o[10] = (__bf16)c10; o[11] = (__bf16)c11;
      o[12] = (__bf16)c12; o[13] = (__bf16)c13; o[14] = (__bf16)0.f; o[15] = (__bf16)0.f;
    }
  }
  copy_lds(sW, gW1, 20480, t);
  if (t < 64) { sRed[t] = gSB[t]; sRed[64 + t] = gSB[64 + t]; }
  __syncthreads();

  // ---- conv1: (32,32,16) -> (30,30,64), K=160, BN+ReLU ----
  conv_wmma<32, 30, 900, 16, 160, 4, true, false>(sIN, sA1, sW, sRed, nullptr, lane, wave);
  __syncthreads();

  // ---- 2x2 maxpool -> (15,15,64); load W2 + SB2 ----
  for (int i = t; i < 15 * 15 * 64; i += THREADS) {
    const int c = i & 63, pxy = i >> 6;
    const int py = pxy / 15, px = pxy % 15;
    const __bf16* q = sA1 + ((2 * py) * 30 + 2 * px) * 64 + c;
    const float m = fmaxf(fmaxf((float)q[0], (float)q[64]),
                          fmaxf((float)q[30 * 64], (float)q[30 * 64 + 64]));
    sPool[i] = (__bf16)m;
  }
  copy_lds(sW, gW2, 73728, t);
  if (t < 64) { sRed[t] = gSB[128 + t]; sRed[64 + t] = gSB[192 + t]; }
  __syncthreads();

  // ---- conv2: (15,15,64) -> (13,13,64), K=576, BN+ReLU ----
  conv_wmma<15, 13, 169, 64, 576, 4, true, false>(sPool, sA2, sW, sRed, nullptr, lane, wave);
  __syncthreads();

  copy_lds(sW, gW3, 73728, t);
  if (t < 64) { sRed[t] = gSB[256 + t]; sRed[64 + t] = gSB[320 + t]; }
  __syncthreads();

  // ---- conv3: (13,13,64) -> (11,11,64), BN+ReLU ----
  conv_wmma<13, 11, 121, 64, 576, 4, true, false>(sA2, sA3, sW, sRed, nullptr, lane, wave);
  __syncthreads();

  copy_lds(sW, gW4, 18432, t);
  __syncthreads();

  // ---- conv4: (11,11,64) -> (9,9,1) f32 ----
  conv_wmma<11, 9, 81, 64, 576, 1, false, true>(sA3, nullptr, sW, nullptr, sS4, lane, wave);
  __syncthreads();

  if (t == 0) {
    float mx = -INFINITY;
    for (int i = 0; i < 81; ++i) mx = fmaxf(mx, sS4[i]);
    out[b] = mx + c4b[0];                        // max(conv+b) == max(conv)+b
  }
}

// ======================================================================
extern "C" void kernel_launch(void* const* d_in, const int* in_sizes, int n_in,
                              void* d_out, int out_size, void* d_ws, size_t ws_size,
                              hipStream_t stream) {
  const float* ts   = (const float*)d_in[0];
  const float* ptm  = (const float*)d_in[1];
  const float* pmt  = (const float*)d_in[2];
  const float* piou = (const float*)d_in[3];
  const int*   msk  = (const int*)d_in[4];
  const float* c1w = (const float*)d_in[5];
  const float* c1b = (const float*)d_in[6];
  const float* g1  = (const float*)d_in[7];
  const float* b1  = (const float*)d_in[8];
  const float* m1  = (const float*)d_in[9];
  const float* v1  = (const float*)d_in[10];
  const float* c2w = (const float*)d_in[11];
  const float* c2b = (const float*)d_in[12];
  const float* g2  = (const float*)d_in[13];
  const float* b2  = (const float*)d_in[14];
  const float* m2  = (const float*)d_in[15];
  const float* v2  = (const float*)d_in[16];
  const float* c3w = (const float*)d_in[17];
  const float* c3b = (const float*)d_in[18];
  const float* g3  = (const float*)d_in[19];
  const float* b3  = (const float*)d_in[20];
  const float* m3  = (const float*)d_in[21];
  const float* v3  = (const float*)d_in[22];
  const float* c4w = (const float*)d_in[23];
  const float* c4b = (const float*)d_in[24];

  char* ws = (char*)d_ws;
  unsigned short* W1 = (unsigned short*)(ws + 0);
  unsigned short* W2 = (unsigned short*)(ws + 20480);
  unsigned short* W3 = (unsigned short*)(ws + 94208);
  unsigned short* W4 = (unsigned short*)(ws + 167936);
  float*          SB = (float*)(ws + 186368);

  prep_kernel<<<(93376 + THREADS - 1) / THREADS, THREADS, 0, stream>>>(
      c1w, c2w, c3w, c4w, c1b, c2b, c3b,
      g1, b1, m1, v1, g2, b2, m2, v2, g3, b3, m3, v3,
      W1, W2, W3, W4, SB);

  fused_kernel<<<512, THREADS, SMEM_BYTES, stream>>>(
      ts, ptm, pmt, piou, msk, W1, W2, W3, W4, SB, c4b, (float*)d_out);

  (void)in_sizes; (void)n_in; (void)out_size; (void)ws_size;
}
